// InvariantPointAttention_88038239633739
// MI455X (gfx1250) — matrix operand order repaired and checked by reading, not compiled, not run on hardware
//
#include <hip/hip_runtime.h>
#include <stdint.h>

// ---------------- problem constants ----------------
constexpr int kB  = 2;
constexpr int kL  = 1024;
constexpr int kC  = 512;
constexpr int kH  = 8;
constexpr int kD  = kC / kH;        // 64
constexpr int kBL = kB * kL;        // 2048
constexpr int kPH3 = kH * 4 * 3;    // 96
constexpr float kLnEps = 1e-5f;

typedef __attribute__((ext_vector_type(16))) _Float16 v16h;
typedef __attribute__((ext_vector_type(8)))  _Float16 v8h;
typedef __attribute__((ext_vector_type(8)))  float    v8f;

// ---------------- WMMA helpers ----------------
__device__ __forceinline__ v8f wmma_f16(v16h a, v16h b, v8f c) {
  return __builtin_amdgcn_wmma_f32_16x16x32_f16(
      /*neg_a=*/false, a, /*neg_b=*/false, b,
      /*c_mod=*/(short)0, c, /*reuse_a=*/false, /*reuse_b=*/false);
}

// A fragment (16x32 f16, MxK), row-major source with leading dim ldh (halves).
// Layout: lanes 0-15 hold K=0..15, lanes 16-31 hold K=16..31 (matches B).
__device__ __forceinline__ v16h load_a16(const _Float16* base, int ldh, int lane) {
  int row = lane & 15;
  int off = (lane >> 4) << 4;                 // 0 or 16
  const _Float16* p = base + (size_t)row * ldh + off;
  union { v16h v; v8h h[2]; } u;
  u.h[0] = *(const v8h*)(p);                  // K = off .. off+7
  u.h[1] = *(const v8h*)(p + 8);              // K = off+8 .. off+15
  return u.v;
}

// B fragment (32x16 f16, KxN) from an N-major [N,K] source (leading dim ldh).
// Layout used here (contiguous k, same as A): lanes 0-15 hold K=0..15, lanes 16-31 hold K=16..31; N = lane%16.
__device__ __forceinline__ v16h load_b16(const _Float16* base, int ldh, int lane) {
  int n    = lane & 15;
  int koff = (lane >> 4) << 4;                // 0 or 16
  const _Float16* p = base + (size_t)n * ldh + koff;
  union { v16h v; v8h h[2]; } u;
  u.h[0] = *(const v8h*)(p);
  u.h[1] = *(const v8h*)(p + 8);
  return u.v;
}

// B fragment from K-major [K,N] source (row stride in halves): strided reads.
__device__ __forceinline__ v16h load_b16_strided(const _Float16* base, int stride, int lane) {
  int n    = lane & 15;
  int koff = (lane >> 4) << 4;
  v16h v;
#pragma unroll
  for (int e = 0; e < 16; ++e)
    v[e] = base[(size_t)(koff + e) * stride + n];
  return v;
}

// ---------------- CDNA5 async copy: global -> LDS, 16B per lane ----------------
__device__ __forceinline__ void async_copy_b128(uint32_t lds_byte, const void* gaddr) {
  asm volatile("global_load_async_to_lds_b128 %0, %1, off"
               :: "v"(lds_byte), "v"(gaddr) : "memory");
}
__device__ __forceinline__ void wait_async0() {
  asm volatile("s_wait_asynccnt 0x0" ::: "memory");
}

// ---------------- 1) LayerNorm -> f16 ----------------
__global__ void __launch_bounds__(256)
ln_f16_kernel(const float* __restrict__ x, const float* __restrict__ g,
              const float* __restrict__ bb, _Float16* __restrict__ y) {
  int row = blockIdx.x;
  const float* xr = x + (size_t)row * kC;
  int t = threadIdx.x;
  float v0 = xr[t], v1 = xr[t + 256];
  float s = v0 + v1, s2 = v0 * v0 + v1 * v1;
#pragma unroll
  for (int m = 16; m >= 1; m >>= 1) {
    s  += __shfl_xor(s,  m, 32);
    s2 += __shfl_xor(s2, m, 32);
  }
  __shared__ float sh[2][8];
  int w = t >> 5, lane = t & 31;
  if (lane == 0) { sh[0][w] = s; sh[1][w] = s2; }
  __syncthreads();
  if (w == 0) {
    float a  = (lane < 8) ? sh[0][lane] : 0.f;
    float b2 = (lane < 8) ? sh[1][lane] : 0.f;
#pragma unroll
    for (int m = 4; m >= 1; m >>= 1) {
      a  += __shfl_xor(a,  m, 32);
      b2 += __shfl_xor(b2, m, 32);
    }
    if (lane == 0) { sh[0][0] = a; sh[1][0] = b2; }
  }
  __syncthreads();
  float mean = sh[0][0] * (1.f / kC);
  float var  = sh[1][0] * (1.f / kC) - mean * mean;
  float rs   = rsqrtf(var + kLnEps);
  y[(size_t)row * kC + t]       = (_Float16)((v0 - mean) * rs * g[t]       + bb[t]);
  y[(size_t)row * kC + t + 256] = (_Float16)((v1 - mean) * rs * g[t + 256] + bb[t + 256]);
}

// ---------------- 2) weight convert+transpose: [K,N] f32 -> [N,K] f16 ----
__global__ void __launch_bounds__(256)
wt_convert_kernel(const float* __restrict__ w, _Float16* __restrict__ wt,
                  int K, int N) {
  size_t idx = (size_t)blockIdx.x * 256 + threadIdx.x;
  if (idx >= (size_t)K * N) return;
  int n = (int)(idx % N);
  int k = (int)(idx / N);
  wt[(size_t)n * K + k] = (_Float16)w[idx];     // coalesced read, strided write
}

// ---------------- 3) WMMA GEMM with bias: out = A(f16)[M,K] * BT(f16)[N,K]^T + bias
template <int NF, bool HOUT>
__global__ void __launch_bounds__(128)
gemm_bias_kernel(const _Float16* __restrict__ A, const _Float16* __restrict__ BT,
                 const float* __restrict__ bias, void* __restrict__ out,
                 int M, int N, int K) {
  int lane = threadIdx.x & 31;
  int wave = threadIdx.x >> 5;
  int m0 = (blockIdx.x * 4 + wave) * 16;
  int n0 = blockIdx.y * (NF * 16);
  if (m0 >= M) return;

  v8f acc[NF] = {};
  for (int k0 = 0; k0 < K; k0 += 32) {
    // prefetch next K-slice of A (one row per lane half, L2 -> WGP)
    if (k0 + 32 < K)
      __builtin_prefetch(A + (size_t)(m0 + (lane & 15)) * K + k0 + 32, 0, 1);
    v16h a = load_a16(A + (size_t)m0 * K + k0, K, lane);
#pragma unroll
    for (int f = 0; f < NF; ++f) {
      v16h b = load_b16(BT + (size_t)(n0 + f * 16) * K + k0, K, lane);
      acc[f] = wmma_f16(a, b, acc[f]);
    }
  }
  int hi8 = (lane >> 4) << 3;
  int nl  = lane & 15;
#pragma unroll
  for (int f = 0; f < NF; ++f) {
    int n = n0 + f * 16 + nl;
    float bv = bias[n];
#pragma unroll
    for (int r = 0; r < 8; ++r) {
      int m = m0 + r + hi8;
      float val = acc[f][r] + bv;
      if (HOUT) ((_Float16*)out)[(size_t)m * N + n] = (_Float16)val;
      else      ((float*)out)[(size_t)m * N + n] = val;
    }
  }
}

// ---------------- 4) |q_p|^2 / |k_p|^2 per (b,h,l) ----------------
__global__ void __launch_bounds__(256)
psq_kernel(const float* __restrict__ qp, const float* __restrict__ kp,
           float* __restrict__ qsq, float* __restrict__ ksq) {
  int idx = blockIdx.x * 256 + threadIdx.x;   // over BL*H
  if (idx >= kBL * kH) return;
  int bl = idx / kH, h = idx % kH;
  const float* q = qp + (size_t)bl * kPH3 + h * 12;
  const float* k = kp + (size_t)bl * kPH3 + h * 12;
  float sq = 0.f, sk = 0.f;
#pragma unroll
  for (int t = 0; t < 12; ++t) { sq += q[t] * q[t]; sk += k[t] * k[t]; }
  int b = bl / kL, i = bl % kL;
  size_t o = ((size_t)b * kH + h) * kL + i;
  qsq[o] = sq;
  ksq[o] = sk;
}

// ---------------- 5) flash attention with point-distance bias ----------------
// One wave per 16-row i-tile of one (b,h). Per 32-wide j-block:
//   - async-copy V tile (32x64 f16) global -> wave-private LDS (overlapped)
//   - S = Q*K^T via 4 WMMAs, point bias + online softmax in f32 VALU
//   - P staged through LDS (C-layout -> A-layout), O += P*V via 4 WMMAs
__global__ void __launch_bounds__(128)
ipa_attn_kernel(const _Float16* __restrict__ q16, const _Float16* __restrict__ k16,
                const _Float16* __restrict__ v16f,
                const float* __restrict__ kp,
                const float* __restrict__ qp,
                const float* __restrict__ qsq, const float* __restrict__ ksq,
                const float* __restrict__ pscale,
                _Float16* __restrict__ out16) {
  __shared__ __align__(16) _Float16 p_st[4][16][32];
  __shared__ __align__(16) _Float16 v_st[4][32][64];
  __shared__ float qp_st[4][16][12];
  __shared__ float qs_st[4][16];

  int lane = threadIdx.x & 31;
  int w    = threadIdx.x >> 5;
  int g    = blockIdx.x * 4 + w;
  constexpr int ntiles = kL / 16;             // 64
  int it = g % ntiles;
  int h  = (g / ntiles) % kH;
  int b  = g / (ntiles * kH);
  int i0 = it * 16;
  size_t rowq = (size_t)b * kL + i0;

  // Stage q_p rows + |q_p|^2 (wave-private LDS; DS is in-order per wave)
  if (lane < 16) {
    const float* src = qp + (rowq + lane) * kPH3 + h * 12;
#pragma unroll
    for (int t = 0; t < 12; ++t) qp_st[w][lane][t] = src[t];
    qs_st[w][lane] = qsq[((size_t)b * kH + h) * kL + i0 + lane];
  }

  const _Float16* qbase = q16 + rowq * kC + h * kD;
  v16h aq0 = load_a16(qbase,      kC, lane);  // d = 0..31
  v16h aq1 = load_a16(qbase + 32, kC, lane);  // d = 32..63

  float ps = pscale[h];
  float mrow[8], srow[8];
#pragma unroll
  for (int r = 0; r < 8; ++r) { mrow[r] = -1e30f; srow[r] = 0.f; }
  v8f o[4] = {};

  int hi8 = (lane >> 4) << 3;
  int nl  = lane & 15;
  const float ssc = 0.125f;                   // 1/sqrt(D), D=64
  const float* ksqh = ksq + ((size_t)b * kH + h) * kL;
  const uint32_t vst_base = (uint32_t)(uintptr_t)&v_st[w][0][0];

  for (int jb = 0; jb < kL; jb += 32) {
    // ---- kick off async V tile copy (32 rows x 64 halves = 256 x 16B chunks) ----
    {
      const _Float16* vbase = v16f + ((size_t)b * kL + jb) * kC + h * kD;
#pragma unroll
      for (int c = 0; c < 8; ++c) {
        int chunk = c * 32 + lane;            // 0..255
        int row = chunk >> 3, seg = chunk & 7;
        async_copy_b128(vst_base + (uint32_t)(row * 64 + seg * 8) * 2,
                        vbase + (size_t)row * kC + seg * 8);
      }
    }
    // prefetch next j-block's K rows (one row per lane) into WGP/L2
    if (jb + 32 < kL)
      __builtin_prefetch(k16 + ((size_t)b * kL + jb + 32 + lane) * kC + h * kD, 0, 1);

    const _Float16* kbase = k16 + ((size_t)b * kL + jb) * kC + h * kD;
    v16h bk00 = load_b16(kbase,                kC, lane);   // j 0..15,  d 0..31
    v16h bk01 = load_b16(kbase + 32,           kC, lane);   // j 0..15,  d 32..63
    v16h bk10 = load_b16(kbase + 16 * kC,      kC, lane);
    v16h bk11 = load_b16(kbase + 16 * kC + 32, kC, lane);
    v8f s0 = {}, s1 = {};
    s0 = wmma_f16(aq0, bk00, s0);  s0 = wmma_f16(aq1, bk01, s0);
    s1 = wmma_f16(aq0, bk10, s1);  s1 = wmma_f16(aq1, bk11, s1);

    // point-distance bias (rank-12, f32 VALU)
    int j0 = jb + nl, j1 = jb + 16 + nl;
    const float* kp0 = kp + ((size_t)b * kL + j0) * kPH3 + h * 12;
    const float* kp1 = kp + ((size_t)b * kL + j1) * kPH3 + h * 12;
    float kpl0[12], kpl1[12];
#pragma unroll
    for (int t = 0; t < 12; ++t) { kpl0[t] = kp0[t]; kpl1[t] = kp1[t]; }
    float ks0 = ksqh[j0], ks1 = ksqh[j1];

    float l0[8], l1[8];
#pragma unroll
    for (int r = 0; r < 8; ++r) {
      int M = r + hi8;
      float d0 = 0.f, d1 = 0.f;
#pragma unroll
      for (int t = 0; t < 12; ++t) {
        float qv = qp_st[w][M][t];
        d0 = fmaf(qv, kpl0[t], d0);
        d1 = fmaf(qv, kpl1[t], d1);
      }
      float qs = qs_st[w][M];
      l0[r] = s0[r] * ssc - ps * (qs + ks0 - 2.f * d0);
      l1[r] = s1[r] * ssc - ps * (qs + ks1 - 2.f * d1);
    }

    // online softmax: rows live across 16-lane halves -> xor-reduce masks 8..1
#pragma unroll
    for (int r = 0; r < 8; ++r) {
      float mx = fmaxf(l0[r], l1[r]);
#pragma unroll
      for (int msk = 8; msk >= 1; msk >>= 1) mx = fmaxf(mx, __shfl_xor(mx, msk, 32));
      float mnew  = fmaxf(mrow[r], mx);
      float scale = __expf(mrow[r] - mnew);
      float p0 = __expf(l0[r] - mnew);
      float p1 = __expf(l1[r] - mnew);
      float psum = p0 + p1;
#pragma unroll
      for (int msk = 8; msk >= 1; msk >>= 1) psum += __shfl_xor(psum, msk, 32);
      srow[r] = srow[r] * scale + psum;
      mrow[r] = mnew;
#pragma unroll
      for (int f = 0; f < 4; ++f) o[f][r] *= scale;
      int M = r + hi8;
      p_st[w][M][nl]      = (_Float16)p0;
      p_st[w][M][16 + nl] = (_Float16)p1;
    }

    // C-layout -> A-layout through wave-private LDS tile
    v16h ap = load_a16(&p_st[w][0][0], 32, lane);

    // async V tile must be resident; then B-frags via strided LDS reads
    wait_async0();
#pragma unroll
    for (int f = 0; f < 4; ++f) {
      v16h bv = load_b16_strided(&v_st[w][0][f * 16], 64, lane);
      o[f] = wmma_f16(ap, bv, o[f]);
    }
  }

  // normalize and emit f16 for the output projection GEMM
#pragma unroll
  for (int r = 0; r < 8; ++r) {
    float inv = 1.f / srow[r];
    int M = r + hi8;
    size_t orow = (rowq + M) * kC + h * kD;
#pragma unroll
    for (int f = 0; f < 4; ++f)
      out16[orow + f * 16 + nl] = (_Float16)(o[f][r] * inv);
  }
}

// ---------------- host-side orchestration ----------------
extern "C" void kernel_launch(void* const* d_in, const int* in_sizes, int n_in,
                              void* d_out, int out_size, void* d_ws, size_t ws_size,
                              hipStream_t stream) {
  (void)in_sizes; (void)n_in; (void)out_size; (void)ws_size;
  const float* features = (const float*)d_in[0];
  // d_in[1] = coords (unused by reference)
  const float* ln_g = (const float*)d_in[2];
  const float* ln_b = (const float*)d_in[3];
  const float* wq  = (const float*)d_in[4];
  const float* bq  = (const float*)d_in[5];
  const float* wk  = (const float*)d_in[6];
  const float* bk  = (const float*)d_in[7];
  const float* wv  = (const float*)d_in[8];
  const float* bv  = (const float*)d_in[9];
  const float* wqp = (const float*)d_in[10];
  const float* bqp = (const float*)d_in[11];
  const float* wkp = (const float*)d_in[12];
  const float* bkp = (const float*)d_in[13];
  // wvp/bvp (14/15) unused by reference
  const float* wo  = (const float*)d_in[16];
  const float* bo  = (const float*)d_in[17];
  const float* pscale = (const float*)d_in[18];

  uint8_t* wsp = (uint8_t*)d_ws;
  auto alloc = [&](size_t bytes) -> void* {
    void* r = (void*)wsp;
    wsp += (bytes + 255) & ~(size_t)255;
    return r;
  };
  _Float16* x16    = (_Float16*)alloc((size_t)kBL * kC * 2);
  _Float16* wq16   = (_Float16*)alloc((size_t)kC * kC * 2);
  _Float16* wk16   = (_Float16*)alloc((size_t)kC * kC * 2);
  _Float16* wv16   = (_Float16*)alloc((size_t)kC * kC * 2);
  _Float16* wo16   = (_Float16*)alloc((size_t)kC * kC * 2);
  _Float16* wqp16  = (_Float16*)alloc((size_t)kC * kPH3 * 2);
  _Float16* wkp16  = (_Float16*)alloc((size_t)kC * kPH3 * 2);
  _Float16* q16    = (_Float16*)alloc((size_t)kBL * kC * 2);
  _Float16* k16    = (_Float16*)alloc((size_t)kBL * kC * 2);
  _Float16* v16b   = (_Float16*)alloc((size_t)kBL * kC * 2);
  _Float16* attn16 = (_Float16*)alloc((size_t)kBL * kC * 2);
  float*    qp     = (float*)alloc((size_t)kBL * kPH3 * 4);
  float*    kp     = (float*)alloc((size_t)kBL * kPH3 * 4);
  float*    qsq    = (float*)alloc((size_t)kB * kH * kL * 4);
  float*    ksq    = (float*)alloc((size_t)kB * kH * kL * 4);

  // 1) LN -> f16
  ln_f16_kernel<<<kBL, 256, 0, stream>>>(features, ln_g, ln_b, x16);

  // 2) weights -> f16 [N,K]
  wt_convert_kernel<<<(kC * kC + 255) / 256, 256, 0, stream>>>(wq,  wq16,  kC, kC);
  wt_convert_kernel<<<(kC * kC + 255) / 256, 256, 0, stream>>>(wk,  wk16,  kC, kC);
  wt_convert_kernel<<<(kC * kC + 255) / 256, 256, 0, stream>>>(wv,  wv16,  kC, kC);
  wt_convert_kernel<<<(kC * kC + 255) / 256, 256, 0, stream>>>(wo,  wo16,  kC, kC);
  wt_convert_kernel<<<(kC * kPH3 + 255) / 256, 256, 0, stream>>>(wqp, wqp16, kC, kPH3);
  wt_convert_kernel<<<(kC * kPH3 + 255) / 256, 256, 0, stream>>>(wkp, wkp16, kC, kPH3);

  // 3) scalar q/k/v projections (f16 output for attention WMMAs)
  dim3 gC(kBL / 64, kC / 64);
  gemm_bias_kernel<4, true><<<gC, 128, 0, stream>>>(x16, wq16, bq, q16,  kBL, kC, kC);
  gemm_bias_kernel<4, true><<<gC, 128, 0, stream>>>(x16, wk16, bk, k16,  kBL, kC, kC);
  gemm_bias_kernel<4, true><<<gC, 128, 0, stream>>>(x16, wv16, bv, v16b, kBL, kC, kC);

  // 4) point q/k projections (f32 output, N=96)
  dim3 gP(kBL / 64, 3);
  gemm_bias_kernel<2, false><<<gP, 128, 0, stream>>>(x16, wqp16, bqp, qp, kBL, kPH3, kC);
  gemm_bias_kernel<2, false><<<gP, 128, 0, stream>>>(x16, wkp16, bkp, kp, kBL, kPH3, kC);

  // 5) |q_p|^2, |k_p|^2
  psq_kernel<<<(kBL * kH + 255) / 256, 256, 0, stream>>>(qp, kp, qsq, ksq);

  // 6) flash attention with point-distance bias (async V staging)
  ipa_attn_kernel<<<(kB * kH * (kL / 16)) / 4, 128, 0, stream>>>(
      q16, k16, v16b, kp, qp, qsq, ksq, pscale, attn16);

  // 7) output projection -> f32 d_out
  gemm_bias_kernel<4, false><<<gC, 128, 0, stream>>>(attn16, wo16, bo, (float*)d_out,
                                                     kBL, kC, kC);
}
